// LSTMEncoder_65515431133655
// MI455X (gfx1250) — matrix operand order repaired and checked by reading, not compiled, not run on hardware
//
#include <hip/hip_runtime.h>
#include <hip/hip_bf16.h>

// ---------------- problem constants ----------------
#define Bn 16
#define Ln 96
#define Hn 150
#define M_ROWS (Bn * Ln)   // 1536 (b*L + t) rows
#define KP 320             // padded K for all GEMMs (>=300, mult of 32)
#define NG 640             // padded gate dim: 4 gates x 160
#define HP 160             // padded hidden (150 -> 160)

typedef __attribute__((ext_vector_type(16))) _Float16 v16h_t;
typedef __attribute__((ext_vector_type(8)))  _Float16 v8h_t;
typedef __attribute__((ext_vector_type(8)))  float    v8f_t;

// ---- WMMA fragment loaders (layouts per CDNA5 ISA 7.12.2) ----
// A 16x32 f16: lane l -> row m=l&15; halves = K[kh*8 .. +7] and K[16+kh*8 .. +7]
__device__ __forceinline__ v16h_t load_a_frag(const _Float16* A, int lda, int ks) {
  int l = threadIdx.x & 31;
  int m = l & 15;
  int kh = l >> 4;
  union { v16h_t v; v8h_t h[2]; } u;
  const _Float16* p = A + (long)m * lda + ks + kh * 8;
  u.h[0] = *(const v8h_t*)(p);
  u.h[1] = *(const v8h_t*)(p + 16);
  return u.v;
}
// B 32x16 f16 (we feed B^T = W row-major NxK): lane l -> col n=l&15,
// halves = K[kh*8 .. +7] and K[16+kh*8 .. +7] of W row n (same as A).
__device__ __forceinline__ v16h_t load_b_frag(const _Float16* W, int ldb, int ks) {
  int l = threadIdx.x & 31;
  int n = l & 15;
  int kh = l >> 4;
  union { v16h_t v; v8h_t h[2]; } u;
  const _Float16* p = W + (long)n * ldb + ks + kh * 8;
  u.h[0] = *(const v8h_t*)(p);
  u.h[1] = *(const v8h_t*)(p + 16);
  return u.v;
}

// ---------------- data prep kernels ----------------
__global__ void embed_kernel(const int* __restrict__ widx, const int* __restrict__ pidx,
                             const float* __restrict__ wemb, const float* __restrict__ temb,
                             _Float16* __restrict__ x) {
  long idx = (long)blockIdx.x * blockDim.x + threadIdx.x;
  if (idx >= (long)M_ROWS * KP) return;
  int m = (int)(idx / KP), k = (int)(idx % KP);
  float v = 0.f;
  if (k < 100)      v = wemb[(long)widx[m] * 100 + k];
  else if (k < 150) v = temb[(long)pidx[m] * 50 + (k - 100)];
  x[idx] = (_Float16)v;
}

// Wih -> f16, gate-padded (640 x 320) per (layer, dir)
__global__ void conv_wih_kernel(const float* __restrict__ W_ih0,
                                const float* __restrict__ W_ih_rest,
                                _Float16* __restrict__ Wihp) {
  long idx = (long)blockIdx.x * blockDim.x + threadIdx.x;
  if (idx >= 8L * NG * KP) return;
  int ld = (int)(idx / ((long)NG * KP));
  long r = idx % ((long)NG * KP);
  int np = (int)(r / KP), k = (int)(r % KP);
  int layer = ld >> 1, d = ld & 1;
  int g = np / HP, jj = np % HP;
  float v = 0.f;
  if (jj < Hn) {
    int row = g * Hn + jj;
    if (layer == 0) { if (k < 150) v = W_ih0[((long)d * 600 + row) * 150 + k]; }
    else            { if (k < 300) v = W_ih_rest[(((long)(layer - 1) * 2 + d) * 600 + row) * 300 + k]; }
  }
  Wihp[idx] = (_Float16)v;
}

// Whh -> f16 gate-padded (640 x 160) per (layer, dir); stays L2-resident
__global__ void conv_whh_kernel(const float* __restrict__ W_hh, _Float16* __restrict__ Whhp) {
  long idx = (long)blockIdx.x * blockDim.x + threadIdx.x;
  if (idx >= 8L * NG * HP) return;
  int ld = (int)(idx / (NG * HP));
  int r = (int)(idx % (NG * HP));
  int np = r / HP, k = r % HP;
  int g = np / HP, jj = np % HP;
  float v = 0.f;
  if (jj < Hn && k < Hn) v = W_hh[((long)ld * 600 + g * Hn + jj) * Hn + k];
  Whhp[idx] = (_Float16)v;
}

// bias = b_ih + b_hh, gate-padded (8 x 640)
__global__ void conv_bias_kernel(const float* __restrict__ b_ih, const float* __restrict__ b_hh,
                                 float* __restrict__ biasp) {
  int idx = blockIdx.x * blockDim.x + threadIdx.x;
  if (idx >= 8 * NG) return;
  int ld = idx / NG, np = idx % NG;
  int g = np / HP, jj = np % HP;
  float v = 0.f;
  if (jj < Hn) { int row = g * Hn + jj; v = b_ih[ld * 600 + row] + b_hh[ld * 600 + row]; }
  biasp[idx] = v;
}

// Stack W1a = mlp_w1[:, :300] (rows 0..99) and W1b = mlp_w1[:, 300:] (rows 100..199),
// padded to 208 x 320 f16. bias2[r] = b1[r] for r<100 else 0.
__global__ void conv_mlp_kernel(const float* __restrict__ mlp_w1, const float* __restrict__ mlp_b1,
                                _Float16* __restrict__ Wm, float* __restrict__ bias2) {
  int idx = blockIdx.x * blockDim.x + threadIdx.x;
  if (idx < 208 * KP) {
    int r = idx / KP, k = idx % KP;
    float v = 0.f;
    if (k < 300) {
      if (r < 100)      v = mlp_w1[r * 600 + k];
      else if (r < 200) v = mlp_w1[(r - 100) * 600 + 300 + k];
    }
    Wm[idx] = (_Float16)v;
  }
  if (idx < 208) bias2[idx] = (idx < 100) ? mlp_b1[idx] : 0.f;
}

// ---------------- generic WMMA GEMM: C(MxN) = A_f16(MxKp) @ W_f16(NxKp)^T + bias[N]
// one wave per 16x16 output tile; batched over blockIdx.z
__global__ void gemm_wmma_kernel(const _Float16* __restrict__ A, int lda,
                                 const _Float16* __restrict__ W, long strideW, int ldb,
                                 float* __restrict__ C, long strideC, int ldc,
                                 const float* __restrict__ bias, long strideBias,
                                 int Kp) {
  int nt = blockIdx.x, mt = blockIdx.y, z = blockIdx.z;
  const _Float16* Wz = W + (long)z * strideW;
  float* Cz = C + (long)z * strideC;
  const float* bz = bias + (long)z * strideBias;
  int l = threadIdx.x;
  int n = l & 15, kh = l >> 4;
  const _Float16* Arow = A + (long)(mt * 16) * lda;
  const _Float16* Wrow = Wz + (long)(nt * 16) * ldb;
  v8f_t acc = {};
  for (int ks = 0; ks < Kp; ks += 32) {
    v16h_t a = load_a_frag(Arow, lda, ks);
    v16h_t b = load_b_frag(Wrow, ldb, ks);
    acc = __builtin_amdgcn_wmma_f32_16x16x32_f16(false, a, false, b, (short)0, acc, false, false);
  }
  float bv = bz[nt * 16 + n];
  float* cp = Cz + (long)(mt * 16 + kh * 8) * ldc + nt * 16 + n;
#pragma unroll
  for (int i = 0; i < 8; ++i) cp[(long)i * ldc] = acc[i] + bv;
}

// ---------------- sequential LSTM scan, one block per direction ----------------
// h,c in LDS; per step: gates(16x640) = pre + h @ Whh^T via WMMA (40 N-tiles x 5 K-frags),
// then elementwise cell update. d=1 scans time in reverse (== input reversal + output re-reversal).
__global__ void __launch_bounds__(256) lstm_scan_kernel(
    const float* __restrict__ pre,      // (2, 1536, 640)
    const _Float16* __restrict__ Whhp,  // (8, 640, 160) f16 gate-padded
    int layer, int lastLayer,
    _Float16* __restrict__ xnext,       // (1536, 320) f16 — next layer input
    float* __restrict__ outbuf) {       // (1536, 300) f32 — final LSTM output
  __shared__ _Float16 sH[Bn * HP];
  __shared__ float    sC[Bn * HP];
  __shared__ float    sG[Bn * NG];
  int d = blockIdx.x;
  int tid = threadIdx.x;
  for (int idx = tid; idx < Bn * HP; idx += 256) { sH[idx] = (_Float16)0.f; sC[idx] = 0.f; }
  __syncthreads();

  const float* preD = pre + (long)d * M_ROWS * NG;
  const _Float16* Wd = Whhp + (long)(layer * 2 + d) * NG * HP;
  int wave = tid >> 5, l = tid & 31, n = l & 15, kh = l >> 4;

  for (int step = 0; step < Ln; ++step) {
    int t = d ? (Ln - 1 - step) : step;
    // phase 1: gates via WMMA
    v16h_t aF[5];
#pragma unroll
    for (int kk = 0; kk < 5; ++kk) aF[kk] = load_a_frag(sH, HP, kk * 32);
    for (int tile = wave; tile < 40; tile += 8) {
      const _Float16* Wrow = Wd + (long)(tile * 16) * HP;
      v8f_t acc;
#pragma unroll
      for (int i = 0; i < 8; ++i) {
        int b = kh * 8 + i;
        acc[i] = preD[((long)(b * Ln + t)) * NG + tile * 16 + n];
      }
#pragma unroll
      for (int kk = 0; kk < 5; ++kk) {
        v16h_t bf = load_b_frag(Wrow, HP, kk * 32);
        acc = __builtin_amdgcn_wmma_f32_16x16x32_f16(false, aF[kk], false, bf, (short)0, acc, false, false);
      }
      int g = tile / 10;  // 0:i 1:f 2:g 3:o
#pragma unroll
      for (int i = 0; i < 8; ++i) {
        int b = kh * 8 + i;
        float v = acc[i];
        v = (g == 2) ? tanhf(v) : (1.0f / (1.0f + expf(-v)));
        sG[b * NG + tile * 16 + n] = v;
      }
    }
    __syncthreads();
    // phase 2: cell update
    for (int idx = tid; idx < Bn * Hn; idx += 256) {
      int b = idx / Hn, j = idx - (idx / Hn) * Hn;
      float ig = sG[b * NG + j];
      float fg = sG[b * NG + HP + j];
      float gg = sG[b * NG + 2 * HP + j];
      float og = sG[b * NG + 3 * HP + j];
      float c = fg * sC[b * HP + j] + ig * gg;
      float h = og * tanhf(c);
      sC[b * HP + j] = c;
      sH[b * HP + j] = (_Float16)h;
      long row = (long)b * Ln + t;
      xnext[row * KP + d * Hn + j] = (_Float16)h;
      if (lastLayer) outbuf[row * 300 + d * Hn + j] = h;
    }
    __syncthreads();
  }
}

// ---------------- elementwise quartic poly ----------------
__global__ void poly_kernel(const float* __restrict__ outbuf, _Float16* __restrict__ P,
                            const float* __restrict__ w1, const float* __restrict__ w2,
                            const float* __restrict__ w3, const float* __restrict__ w4) {
  long idx = (long)blockIdx.x * blockDim.x + threadIdx.x;
  if (idx >= (long)M_ROWS * KP) return;
  int m = (int)(idx / KP), k = (int)(idx % KP);
  float p = 0.f;
  if (k < 300) {
    float v = outbuf[(long)m * 300 + k];
    p = v * (w1[0] + v * (w2[0] + v * (w3[0] + v * w4[0])));
  }
  P[idx] = (_Float16)p;
}

// ---------------- combine: score[lh,b,ld] = b2 + w2 . tanh(U[b,lh] + V[b,ld]) ----------------
__global__ void __launch_bounds__(256) combine_kernel(
    const float* __restrict__ UV,   // (1536, 208): cols 0..99 = U, 100..199 = V
    const float* __restrict__ w2,   // (100)
    const float* __restrict__ b2,   // (1)
    float* __restrict__ out) {      // (96, 16, 96)
  __shared__ float sU[100];
  __shared__ float sw2[100];
  int blk = blockIdx.x;             // b*96 + lh
  int b = blk / Ln, lh = blk % Ln;
  int tid = threadIdx.x;
  const float* Urow = UV + (long)(b * Ln + lh) * 208;
  if (tid < 100) { sU[tid] = Urow[tid]; sw2[tid] = w2[tid]; }
  __syncthreads();
  float bias = b2[0];
  int wave = tid >> 5, lane = tid & 31;
  for (int ld = wave; ld < Ln; ld += 8) {
    const float* Vrow = UV + (long)(b * Ln + ld) * 208 + 100;
    float s = 0.f;
    for (int j = lane; j < 100; j += 32) s += sw2[j] * tanhf(sU[j] + Vrow[j]);
#pragma unroll
    for (int off = 16; off > 0; off >>= 1) s += __shfl_down(s, off, 32);
    if (lane == 0) out[((long)lh * Bn + b) * Ln + ld] = s + bias;
  }
}

extern "C" void kernel_launch(void* const* d_in, const int* in_sizes, int n_in,
                              void* d_out, int out_size, void* d_ws, size_t ws_size,
                              hipStream_t stream) {
  const int*   widx      = (const int*)d_in[0];
  const int*   pidx      = (const int*)d_in[1];
  const float* wemb      = (const float*)d_in[3];
  const float* temb      = (const float*)d_in[4];
  const float* W_ih0     = (const float*)d_in[5];
  const float* W_ih_rest = (const float*)d_in[6];
  const float* W_hh      = (const float*)d_in[7];
  const float* b_ih      = (const float*)d_in[8];
  const float* b_hh      = (const float*)d_in[9];
  const float* w1        = (const float*)d_in[10];
  const float* w2        = (const float*)d_in[11];
  const float* w3        = (const float*)d_in[12];
  const float* w4        = (const float*)d_in[13];
  const float* mlp_w1    = (const float*)d_in[14];
  const float* mlp_b1    = (const float*)d_in[15];
  const float* mlp_w2    = (const float*)d_in[16];
  const float* mlp_b2    = (const float*)d_in[17];

  char* p = (char*)d_ws;
  auto alloc = [&](size_t bytes) { void* r = (void*)p; p += (bytes + 255) & ~(size_t)255; return r; };
  _Float16* x      = (_Float16*)alloc((size_t)M_ROWS * KP * 2);
  _Float16* Wihp   = (_Float16*)alloc((size_t)8 * NG * KP * 2);
  _Float16* Whhp   = (_Float16*)alloc((size_t)8 * NG * HP * 2);
  float*    biasp  = (float*)alloc((size_t)8 * NG * 4);
  float*    pre    = (float*)alloc((size_t)2 * M_ROWS * NG * 4);
  float*    outbuf = (float*)alloc((size_t)M_ROWS * 300 * 4);
  _Float16* P      = (_Float16*)alloc((size_t)M_ROWS * KP * 2);
  _Float16* Wm     = (_Float16*)alloc((size_t)208 * KP * 2);
  float*    bias2  = (float*)alloc((size_t)208 * 4);
  float*    UV     = (float*)alloc((size_t)M_ROWS * 208 * 4);

  long nEmb = (long)M_ROWS * KP;
  embed_kernel<<<dim3((unsigned)((nEmb + 255) / 256)), 256, 0, stream>>>(widx, pidx, wemb, temb, x);
  long nWih = 8L * NG * KP;
  conv_wih_kernel<<<dim3((unsigned)((nWih + 255) / 256)), 256, 0, stream>>>(W_ih0, W_ih_rest, Wihp);
  long nWhh = 8L * NG * HP;
  conv_whh_kernel<<<dim3((unsigned)((nWhh + 255) / 256)), 256, 0, stream>>>(W_hh, Whhp);
  conv_bias_kernel<<<dim3((8 * NG + 255) / 256), 256, 0, stream>>>(b_ih, b_hh, biasp);
  conv_mlp_kernel<<<dim3((208 * KP + 255) / 256), 256, 0, stream>>>(mlp_w1, mlp_b1, Wm, bias2);

  for (int layer = 0; layer < 4; ++layer) {
    dim3 g(NG / 16, M_ROWS / 16, 2);  // batched over direction
    gemm_wmma_kernel<<<g, 32, 0, stream>>>(
        x, KP,
        Wihp + (long)layer * 2 * NG * KP, (long)NG * KP, KP,
        pre, (long)M_ROWS * NG, NG,
        biasp + layer * 2 * NG, NG, KP);
    lstm_scan_kernel<<<2, 256, 0, stream>>>(pre, Whhp, layer, (layer == 3) ? 1 : 0, x, outbuf);
  }

  poly_kernel<<<dim3((unsigned)((nEmb + 255) / 256)), 256, 0, stream>>>(outbuf, P, w1, w2, w3, w4);
  dim3 g2(208 / 16, M_ROWS / 16, 1);
  gemm_wmma_kernel<<<g2, 32, 0, stream>>>(P, KP, Wm, 0, KP, UV, 0, 208, bias2, 0, KP);
  combine_kernel<<<M_ROWS, 256, 0, stream>>>(UV, mlp_w2, mlp_b2, (float*)d_out);
}
